// MambaMixBlock_76261439308439
// MI455X (gfx1250) — matrix operand order repaired and checked by reading, not compiled, not run on hardware
//
#include <hip/hip_runtime.h>
#include <hip/hip_bf16.h>
#include <math.h>

#define D_MODEL 768
#define D_INNER 1536
#define D_STATE 16
#define DT_RANK 96
#define H_FFN   2048
#define SEQ_L   2048
#define EPS_RMS 1e-6f

typedef __attribute__((ext_vector_type(16))) __bf16 v16bf;
typedef __attribute__((ext_vector_type(8)))  float  v8f;

union FragAB { v16bf v; unsigned u[8]; };
union BfPair { __bf16 h[2]; unsigned u; };

// f32 pair -> packed bf16 dword via hardware convert (RNE)
static __device__ __forceinline__ unsigned pack_bf16(float a, float b) {
    BfPair p;
    p.h[0] = (__bf16)a;
    p.h[1] = (__bf16)b;
    return p.u;
}

static __device__ __forceinline__ float softplus_f(float x) {
    return (x > 20.0f) ? x : log1pf(__expf(x));
}
static __device__ __forceinline__ float sigmoid_f(float x) {
    return 1.0f / (1.0f + __expf(-x));
}

// ---------------------------------------------------------------------------
// RMSNorm: one block per row (D elements), out = x * rsqrt(mean(x^2)+eps) * w
// ---------------------------------------------------------------------------
__global__ __launch_bounds__(256) void rmsnorm_kernel(
    const float* __restrict__ x, const float* __restrict__ w,
    float* __restrict__ out, int D) {
    __shared__ float red[256];
    __shared__ float s_scale;
    const int row = blockIdx.x;
    const int tid = threadIdx.x;
    const float* xr = x + (size_t)row * D;
    float acc = 0.0f;
    for (int j = tid; j < D; j += 256) { float v = xr[j]; acc += v * v; }
    red[tid] = acc;
    __syncthreads();
    for (int s = 128; s > 0; s >>= 1) {
        if (tid < s) red[tid] += red[tid + s];
        __syncthreads();
    }
    if (tid == 0) s_scale = rsqrtf(red[0] / (float)D + EPS_RMS);
    __syncthreads();
    float scale = s_scale;
    float* orow = out + (size_t)row * D;
    for (int j = tid; j < D; j += 256) orow[j] = xr[j] * scale * w[j];
}

// ---------------------------------------------------------------------------
// WMMA GEMM:  C[M,N] = A[M,K] @ B[N,K]^T   (both row-major, K contiguous)
// f32 in memory, bf16 tiles in LDS, v_wmma_f32_16x16x32_bf16 compute.
// Block = 256 threads = 8 waves in a 4(M) x 2(N) grid; block tile 64M x 128N;
// each wave computes a 16M x 64N strip = 4 accumulators, reusing its A frag.
// EPI: 0 = none, 1 = softplus(acc + bias[col]), 2 = acc + extra[row*lde+col]
// Requires: M % 64 == 0, N % 128 == 0, K % 32 == 0 (all true here).
// ---------------------------------------------------------------------------
template <int EPI>
__global__ __launch_bounds__(256) void gemm_bf16_wmma(
    const float* __restrict__ A, int lda,
    const float* __restrict__ B, int ldb,
    float* __restrict__ C, int ldc,
    int K, const float* __restrict__ extra, int lde) {
    __shared__ unsigned ldsA[64 * 16];   // 64 rows  x 16 bf16-pairs (K=32) 4KB
    __shared__ unsigned ldsB[128 * 16];  // 128 cols x 16 bf16-pairs (K=32) 8KB

    const int tid  = threadIdx.x;
    const int lane = tid & 31;
    const int wv   = tid >> 5;
    const int wm   = wv >> 1;   // 0..3
    const int wn   = wv & 1;    // 0..1
    const int rowBase = blockIdx.y * 64;
    const int colBase = blockIdx.x * 128;

    v8f acc0 = {}, acc1 = {}, acc2 = {}, acc3 = {};

    const int half = lane >> 4;
    const int l15  = lane & 15;
    const int mL   = wm * 16 + l15;

    for (int kc = 0; kc < K; kc += 32) {
        // Prefetch next K chunk (global_prefetch_b8)
        if (kc + 32 < K) {
            __builtin_prefetch(A + (size_t)(rowBase + (tid >> 3)) * lda + kc + 32, 0, 1);
            __builtin_prefetch(B + (size_t)(colBase + (tid >> 3)) * ldb + kc + 32, 0, 1);
        }
        // Stage A tile: 64 rows x 32 K as float4 -> 2 packed dwords each
        #pragma unroll
        for (int i = 0; i < 2; ++i) {
            int q  = tid + i * 256;          // 0..511
            int r  = q >> 3;                 // row 0..63
            int qc = q & 7;                  // float4 index within K-chunk
            float4 f = *(const float4*)(A + (size_t)(rowBase + r) * lda + kc + qc * 4);
            ldsA[r * 16 + qc * 2 + 0] = pack_bf16(f.x, f.y);
            ldsA[r * 16 + qc * 2 + 1] = pack_bf16(f.z, f.w);
        }
        // Stage B tile: 128 cols x 32 K
        #pragma unroll
        for (int i = 0; i < 4; ++i) {
            int q  = tid + i * 256;          // 0..1023
            int r  = q >> 3;                 // col 0..127
            int qc = q & 7;
            float4 f = *(const float4*)(B + (size_t)(colBase + r) * ldb + kc + qc * 4);
            ldsB[r * 16 + qc * 2 + 0] = pack_bf16(f.x, f.y);
            ldsB[r * 16 + qc * 2 + 1] = pack_bf16(f.z, f.w);
        }
        __syncthreads();

        // A fragment: lane holds row m; K = half*8+[0..7] (v0-3),
        //             K = 16+half*8+[0..7] (v4-7), packed 2 per VGPR.
        FragAB fa;
        #pragma unroll
        for (int r = 0; r < 4; ++r) {
            fa.u[r]     = ldsA[mL * 16 + half * 4 + r];
            fa.u[4 + r] = ldsA[mL * 16 + 8 + half * 4 + r];
        }
        // 4 B fragments; B frag: lane holds col n, K = half*8+[0..7], 16+half*8+[0..7].
        #pragma unroll
        for (int j = 0; j < 4; ++j) {
            FragAB fb;
            const int nL = wn * 64 + j * 16 + l15;
            #pragma unroll
            for (int r = 0; r < 4; ++r) {
                fb.u[r]     = ldsB[nL * 16 + half * 4 + r];
                fb.u[4 + r] = ldsB[nL * 16 + 8 + half * 4 + r];
            }
            v8f c = (j == 0) ? acc0 : (j == 1) ? acc1 : (j == 2) ? acc2 : acc3;
            c = __builtin_amdgcn_wmma_f32_16x16x32_bf16(
                false, fa.v, false, fb.v, (short)0, c, false, false);
            if (j == 0) acc0 = c; else if (j == 1) acc1 = c;
            else if (j == 2) acc2 = c; else acc3 = c;
        }
        __syncthreads();
    }

    // C layout: VGPR v -> M = v + 8*half ; N = lane&15
    #pragma unroll
    for (int j = 0; j < 4; ++j) {
        const v8f acc = (j == 0) ? acc0 : (j == 1) ? acc1 : (j == 2) ? acc2 : acc3;
        const int col = colBase + wn * 64 + j * 16 + l15;
        #pragma unroll
        for (int v = 0; v < 8; ++v) {
            int row = rowBase + wm * 16 + v + 8 * half;
            float val = acc[v];
            if (EPI == 1)      val = softplus_f(val + extra[col]);
            else if (EPI == 2) val += extra[(size_t)row * lde + col];
            C[(size_t)row * ldc + col] = val;
        }
    }
}

// ---------------------------------------------------------------------------
// Elementwise kernels
// ---------------------------------------------------------------------------
__global__ __launch_bounds__(256) void gate_sigmoid_mul_kernel(
    float* __restrict__ z, const float* __restrict__ g, int n) {
    int i = blockIdx.x * 256 + threadIdx.x;
    if (i < n) z[i] = z[i] * sigmoid_f(g[i]);
}

__global__ __launch_bounds__(256) void silu_mul_kernel(
    float* __restrict__ g, const float* __restrict__ u, int n) {
    int i = blockIdx.x * 256 + threadIdx.x;
    if (i < n) { float gv = g[i]; g[i] = gv * sigmoid_f(gv) * u[i]; }
}

// ---------------------------------------------------------------------------
// Selective scan: one thread per channel d; 16-state recurrence in registers;
// per-timestep B/C vectors (xdbl cols 96..127) staged in LDS in 64-t chunks.
// h_t[s] = exp(dt*A[s]) * h_{t-1}[s] + (dt*z)*B_t[s]
// y[t,d] = sum_s h_t[s]*C_t[s] + D_skip[d]*z[t,d]
// ---------------------------------------------------------------------------
__global__ __launch_bounds__(256) void selective_scan_kernel(
    const float* __restrict__ z,    // [L, D_INNER]
    const float* __restrict__ dt,   // [L, D_INNER] (softplus applied)
    const float* __restrict__ xdbl, // [L, 128]; cols 96..111 = B, 112..127 = C
    const float* __restrict__ A_log,// [D_INNER, 16]
    const float* __restrict__ Dsk,  // [D_INNER]
    float* __restrict__ y) {        // [L, D_INNER]
    __shared__ float sBC[64][32];
    const int d = blockIdx.x * 256 + threadIdx.x;   // grid = D_INNER/256

    float a[D_STATE], h[D_STATE];
    #pragma unroll
    for (int s = 0; s < D_STATE; ++s) {
        a[s] = -__expf(A_log[(size_t)d * D_STATE + s]);
        h[s] = 0.0f;
    }
    const float dskip = Dsk[d];

    for (int tc = 0; tc < SEQ_L; tc += 64) {
        __syncthreads();
        for (int i = threadIdx.x; i < 64 * 32; i += 256) {
            int tl = i >> 5, c = i & 31;
            sBC[tl][c] = xdbl[(size_t)(tc + tl) * 128 + 96 + c];
        }
        __syncthreads();
        for (int tl = 0; tl < 64; ++tl) {
            const int t = tc + tl;
            const float dtv = dt[(size_t)t * D_INNER + d];
            const float zv  = z[(size_t)t * D_INNER + d];
            const float dz  = dtv * zv;
            float accv = 0.0f;
            #pragma unroll
            for (int s = 0; s < D_STATE; ++s) {
                float abar = __expf(dtv * a[s]);
                h[s] = abar * h[s] + dz * sBC[tl][s];
                accv += h[s] * sBC[tl][D_STATE + s];
            }
            y[(size_t)t * D_INNER + d] = accv + dskip * zv;
        }
    }
}

// ---------------------------------------------------------------------------
// Launch
// ---------------------------------------------------------------------------
extern "C" void kernel_launch(void* const* d_in, const int* in_sizes, int n_in,
                              void* d_out, int out_size, void* d_ws, size_t ws_size,
                              hipStream_t stream) {
    const float* x          = (const float*)d_in[0];
    const float* w_norm1    = (const float*)d_in[1];
    const float* W_in       = (const float*)d_in[2];
    const float* W_gate     = (const float*)d_in[3];
    const float* W_xproj    = (const float*)d_in[4];
    const float* W_dt       = (const float*)d_in[5];
    const float* b_dt       = (const float*)d_in[6];
    const float* A_log      = (const float*)d_in[7];
    const float* D_skip     = (const float*)d_in[8];
    const float* W_out      = (const float*)d_in[9];
    const float* w_ffn_norm = (const float*)d_in[10];
    const float* W_up       = (const float*)d_in[11];
    const float* W_gate_ffn = (const float*)d_in[12];
    const float* W_down     = (const float*)d_in[13];
    float* out = (float*)d_out;

    float* ws = (float*)d_ws;
    size_t o = 0;
    float* h_norm = ws + o; o += (size_t)SEQ_L * D_MODEL;      // reused as hf
    size_t oz = o;
    float* zbuf   = ws + o; o += (size_t)SEQ_L * D_INNER;      // z_pre -> z
    float* gbuf   = ws + o; o += (size_t)SEQ_L * D_INNER;      // gate_pre -> dt
    float* xdbl   = ws + o; o += (size_t)SEQ_L * 128;
    float* ybuf   = ws + o; o += (size_t)SEQ_L * D_INNER;
    float* x2     = ws + o; o += (size_t)SEQ_L * D_MODEL;
    float* ufbuf  = ws + o; o += (size_t)SEQ_L * H_FFN;
    float* hf     = h_norm;                 // free after in/gate GEMMs
    float* dtbuf  = gbuf;                   // free after sigmoid gating
    float* gfbuf  = ws + oz;                // overlays z+g, free after scan

    const dim3 blk(256);

    // 1) h = rmsnorm(x, w_norm1)
    rmsnorm_kernel<<<SEQ_L, blk, 0, stream>>>(x, w_norm1, h_norm, D_MODEL);

    // 2) z_pre = h @ W_in^T ; 3) gate = h @ W_gate^T
    gemm_bf16_wmma<0><<<dim3(D_INNER / 128, SEQ_L / 64), blk, 0, stream>>>(
        h_norm, D_MODEL, W_in, D_MODEL, zbuf, D_INNER, D_MODEL, nullptr, 0);
    gemm_bf16_wmma<0><<<dim3(D_INNER / 128, SEQ_L / 64), blk, 0, stream>>>(
        h_norm, D_MODEL, W_gate, D_MODEL, gbuf, D_INNER, D_MODEL, nullptr, 0);

    // 4) z = z_pre * sigmoid(gate)
    gate_sigmoid_mul_kernel<<<(SEQ_L * D_INNER) / 256, blk, 0, stream>>>(
        zbuf, gbuf, SEQ_L * D_INNER);

    // 5) xdbl = z @ W_xproj^T   (N = 128)
    gemm_bf16_wmma<0><<<dim3(128 / 128, SEQ_L / 64), blk, 0, stream>>>(
        zbuf, D_INNER, W_xproj, D_INNER, xdbl, 128, D_INNER, nullptr, 0);

    // 6) dt = softplus(xdbl[:, :96] @ W_dt^T + b_dt)   (K = 96, lda = 128)
    gemm_bf16_wmma<1><<<dim3(D_INNER / 128, SEQ_L / 64), blk, 0, stream>>>(
        xdbl, 128, W_dt, DT_RANK, dtbuf, D_INNER, DT_RANK, b_dt, 0);

    // 7) selective scan -> y
    selective_scan_kernel<<<D_INNER / 256, blk, 0, stream>>>(
        zbuf, dtbuf, xdbl, A_log, D_skip, ybuf);

    // 8) x2 = x + y @ W_out^T
    gemm_bf16_wmma<2><<<dim3(D_MODEL / 128, SEQ_L / 64), blk, 0, stream>>>(
        ybuf, D_INNER, W_out, D_INNER, x2, D_MODEL, D_INNER, x, D_MODEL);

    // 9) hf = rmsnorm(x2, w_ffn_norm)
    rmsnorm_kernel<<<SEQ_L, blk, 0, stream>>>(x2, w_ffn_norm, hf, D_MODEL);

    // 10) gf = hf @ W_gate_ffn^T ; 11) uf = hf @ W_up^T
    gemm_bf16_wmma<0><<<dim3(H_FFN / 128, SEQ_L / 64), blk, 0, stream>>>(
        hf, D_MODEL, W_gate_ffn, D_MODEL, gfbuf, H_FFN, D_MODEL, nullptr, 0);
    gemm_bf16_wmma<0><<<dim3(H_FFN / 128, SEQ_L / 64), blk, 0, stream>>>(
        hf, D_MODEL, W_up, D_MODEL, ufbuf, H_FFN, D_MODEL, nullptr, 0);

    // 12) p = silu(gf) * uf   (in place in gf)
    silu_mul_kernel<<<(SEQ_L * H_FFN) / 256, blk, 0, stream>>>(
        gfbuf, ufbuf, SEQ_L * H_FFN);

    // 13) out = x2 + p @ W_down^T
    gemm_bf16_wmma<2><<<dim3(D_MODEL / 128, SEQ_L / 64), blk, 0, stream>>>(
        gfbuf, H_FFN, W_down, H_FFN, out, D_MODEL, H_FFN, x2, D_MODEL);

    (void)in_sizes; (void)n_in; (void)out_size; (void)ws_size;
}
